// TextRNNRepresentation_74062416052856
// MI455X (gfx1250) — matrix operand order repaired and checked by reading, not compiled, not run on hardware
//
#include <hip/hip_runtime.h>
#include <hip/hip_bf16.h>
#include <math.h>

typedef __attribute__((ext_vector_type(16))) _Float16 v16h;
typedef __attribute__((ext_vector_type(8)))  _Float16 v8h;
typedef __attribute__((ext_vector_type(8)))  float    v8f;

#define B_ 64
#define T_ 512
#define E_ 128
#define H_ 256
#define PDEPTH 4   // async prefetch ring depth in the scan kernel

// ---------------------------------------------------------------------------
// A-fragment for V_WMMA_F32_16X16X32_F16 (16-bit A, 16x32, row-major source).
// Per ISA 7.12.2: lanes 0-15 hold K={0..7} in VGPR0-3 and K={16..23} in
// VGPR4-7; lanes 16-31 hold K={8..15} and K={24..31}.  Both halves are
// contiguous 8-half (16B) runs in a row-major matrix.
// ---------------------------------------------------------------------------
__device__ __forceinline__ v16h make_a_frag(const _Float16* rowbase, int half) {
  const v8h lo = *(const v8h*)(rowbase + 8 * half);
  const v8h hi = *(const v8h*)(rowbase + 16 + 8 * half);
  v16h r;
#pragma unroll
  for (int i = 0; i < 8; ++i) { r[i] = lo[i]; r[8 + i] = hi[i]; }
  return r;
}

// B-fragment (32x16): B[k][n] = W[n][k], so lane (n = lane%16, half = lane/16)
// needs W[n0+n][kb*32 + 8*half .. +7] and W[n0+n][kb*32 + 16 + 8*half .. +7].
__device__ __forceinline__ v16h make_b_frag(const _Float16* W, int ld,
                                            int n0, int nl, int half, int kb) {
  const _Float16* p = W + (size_t)(n0 + nl) * ld + kb * 32;
  const v8h lo = *(const v8h*)(p + 8 * half);
  const v8h hi = *(const v8h*)(p + 16 + 8 * half);
  v16h r;
#pragma unroll
  for (int i = 0; i < 8; ++i) { r[i] = lo[i]; r[8 + i] = hi[i]; }
  return r;
}

// Async global->LDS 16B copy (GLOBAL_LOAD_ASYNC_TO_LDS_B128, ASYNCcnt).
// lds_off = low 32 bits of the generic shared pointer (= group-segment offset).
__device__ __forceinline__ void async_g2l_b128(unsigned lds_off, const float* g) {
  asm volatile("global_load_async_to_lds_b128 %0, %1, off"
               :: "v"(lds_off), "v"(g) : "memory");
}
__device__ __forceinline__ void wait_asynccnt_le4() {
  asm volatile("s_wait_asynccnt 0x4" ::: "memory");
}

// ---------------------------------------------------------------------------
// Embedding gather + f32->f16 convert.  One row (128 elems) per block.
// ---------------------------------------------------------------------------
__global__ void gather_kernel(const int* __restrict__ x,
                              const float* __restrict__ emb,
                              _Float16* __restrict__ e16) {
  const int row = blockIdx.x;
  const int k   = threadIdx.x;
  const int idx = x[row];
  e16[(size_t)row * E_ + k] = (_Float16)emb[(size_t)idx * E_ + k];
}

__global__ void cvt_kernel(const float* __restrict__ s,
                           _Float16* __restrict__ d, int n) {
  int i = blockIdx.x * 256 + threadIdx.x;
  if (i < n) d[i] = (_Float16)s[i];
}

// ---------------------------------------------------------------------------
// xproj = X(M,K) @ W(N=256,K).T + (bi + bh), f16 in / f32 out, WMMA.
// One block per 16-row M-tile: the 16xK A-tile (contiguous in memory) is
// staged into LDS once, then 8 waves compute two 16x16 N-tiles each with
// back-to-back WMMAs sharing the A fragment.  gridDim.y picks fwd/bwd.
// ---------------------------------------------------------------------------
__global__ __launch_bounds__(256) void xproj_kernel(
    const _Float16* __restrict__ X, int K,
    const _Float16* __restrict__ Wf, const _Float16* __restrict__ Wb,
    const float* __restrict__ bif, const float* __restrict__ bhf,
    const float* __restrict__ bib, const float* __restrict__ bhb,
    float* __restrict__ outf, float* __restrict__ outb) {
  __shared__ __align__(32) _Float16 xs[16 * 512];  // sized for K=512

  const int tid  = threadIdx.x;
  const int wave = tid >> 5, lane = tid & 31;
  const int nl   = lane & 15, half = lane >> 4;
  const int m0   = blockIdx.x * 16;
  const int dir  = blockIdx.y;

  const _Float16* __restrict__ W  = dir ? Wb  : Wf;
  const float*    __restrict__ bi = dir ? bib : bif;
  const float*    __restrict__ bh = dir ? bhb : bhf;
  float*          __restrict__ out = dir ? outb : outf;

  // Stage A tile: rows m0..m0+15 are one contiguous 16*K run of halfs.
  const v8h* __restrict__ src = (const v8h*)(X + (size_t)m0 * K);
  const int nvec = (16 * K) >> 3;
  for (int i = tid; i < nvec; i += 256) ((v8h*)xs)[i] = src[i];
  __syncthreads();

  const int n0a = wave * 16;
  const int n0b = (wave + 8) * 16;
  v8f acc0 = {0.f, 0.f, 0.f, 0.f, 0.f, 0.f, 0.f, 0.f};
  v8f acc1 = {0.f, 0.f, 0.f, 0.f, 0.f, 0.f, 0.f, 0.f};
  const int K32 = K >> 5;
#pragma unroll 4
  for (int kb = 0; kb < K32; ++kb) {
    v16h a  = make_a_frag(xs + nl * K + kb * 32, half);
    v16h b0 = make_b_frag(W, K, n0a, nl, half, kb);
    v16h b1 = make_b_frag(W, K, n0b, nl, half, kb);
    acc0 = __builtin_amdgcn_wmma_f32_16x16x32_f16(false, a, false, b0,
                                                  (short)0, acc0, false, false);
    acc1 = __builtin_amdgcn_wmma_f32_16x16x32_f16(false, a, false, b1,
                                                  (short)0, acc1, false, false);
  }
  const float biasa = bi[n0a + nl] + bh[n0a + nl];
  const float biasb = bi[n0b + nl] + bh[n0b + nl];
  // C/D layout: lane (n = lane%16, half) holds rows m = 8*half + r in acc[r].
#pragma unroll
  for (int r = 0; r < 8; ++r) {
    const size_t row = (size_t)(m0 + half * 8 + r) * H_;
    out[row + n0a + nl] = acc0[r] + biasa;
    out[row + n0b + nl] = acc1[r] + biasb;
  }
}

// ---------------------------------------------------------------------------
// Recurrent scan: h_t = tanh(xp[t] + h_{t-1} @ Whh.T).
//  * 16 batch rows / block, 16 waves (one 16x16 N-tile each)
//  * Whh fragments resident in VGPRs (loaded once)
//  * h state double-buffered in LDS, one barrier per step
//  * xp[t] tiles streamed PDEPTH-1 steps ahead via GLOBAL_LOAD_ASYNC_TO_LDS_B128
//    into a 4-slot LDS ring; s_wait_asynccnt + the per-step barrier gate reuse
// gridDim = (4 batch-chunks, 2 directions).
// Layer-1 mode: writes h (f16) into the concat sequence buffer o_seq.
// Layer-2 mode: accumulates the T-mean in registers and captures o2[:,T-1,:].
// ---------------------------------------------------------------------------
__global__ __launch_bounds__(512) void rnn_scan_kernel(
    const float* __restrict__ xp_f, const float* __restrict__ xp_b,
    const _Float16* __restrict__ wh_f, const _Float16* __restrict__ wh_b,
    _Float16* __restrict__ o_seq, float* __restrict__ o_last,
    float* __restrict__ o_sum) {
  __shared__ __align__(32) _Float16 hbuf[2][16 * H_];       // 16 KB
  __shared__ __align__(16) float    xpring[PDEPTH][16 * H_]; // 64 KB

  const int tid  = threadIdx.x;
  const int wave = tid >> 5, lane = tid & 31;
  const int nl   = lane & 15, half = lane >> 4;
  const int n0   = wave * 16;
  const int b0   = blockIdx.x * 16;
  const int dir  = blockIdx.y;

  const float*    __restrict__ xp = dir ? xp_b : xp_f;
  const _Float16* __restrict__ wh = dir ? wh_b : wh_f;
  const int col   = dir * H_;
  const int tstep = dir ? -1 : 1;
  const int t0    = dir ? (T_ - 1) : 0;

  // Async-prefetch mapping: thread = (mrow, inner); each thread moves 32B of
  // row (b0+mrow) at time t' via two b128 async copies.
  const int mrow  = tid >> 5;
  const int inner = tid & 31;

  // Whh fragments for this wave's N-tile, all K blocks: 64 VGPRs, loaded once.
  v16h bf[8];
#pragma unroll
  for (int kb = 0; kb < 8; ++kb)
    bf[kb] = make_b_frag(wh, H_, n0, nl, half, kb);

  for (int i = tid; i < 16 * H_; i += 512) hbuf[0][i] = (_Float16)0.f;

  // Prologue: issue steps 0 .. PDEPTH-2 into ring slots 0 .. PDEPTH-2.
#pragma unroll
  for (int p = 0; p < PDEPTH - 1; ++p) {
    const int tt = t0 + p * tstep;
    const float* g = xp + ((size_t)(b0 + mrow) * T_ + tt) * H_ + inner * 8;
    const unsigned l =
        (unsigned)(uintptr_t)&xpring[p][mrow * H_ + inner * 8];
    async_g2l_b128(l, g);
    async_g2l_b128(l + 16u, g + 4);
  }
  __syncthreads();

  float srow[8];
#pragma unroll
  for (int r = 0; r < 8; ++r) srow[r] = 0.f;

  int cur = 0;
  int t = t0;
  for (int s = 0; s < T_; ++s, t += tstep) {
    // Own wave's oldest async pair (step s) has landed; barrier makes every
    // wave's portion of slot s%4 visible, and closes out all reads of slot
    // (s-1)%4 so it can be re-targeted below.
    wait_asynccnt_le4();
    __syncthreads();
    if (s + PDEPTH - 1 < T_) {
      const int sp = s + PDEPTH - 1;
      const int tt = t0 + sp * tstep;
      const float* g = xp + ((size_t)(b0 + mrow) * T_ + tt) * H_ + inner * 8;
      const unsigned l =
          (unsigned)(uintptr_t)&xpring[sp & (PDEPTH - 1)][mrow * H_ + inner * 8];
      async_g2l_b128(l, g);
      async_g2l_b128(l + 16u, g + 4);
    }

    v8f acc = {0.f, 0.f, 0.f, 0.f, 0.f, 0.f, 0.f, 0.f};
#pragma unroll
    for (int kb = 0; kb < 8; ++kb) {
      v16h a = make_a_frag(&hbuf[cur][nl * H_ + kb * 32], half);
      acc = __builtin_amdgcn_wmma_f32_16x16x32_f16(false, a, false, bf[kb],
                                                   (short)0, acc, false, false);
    }
    const float* __restrict__ xs = xpring[s & (PDEPTH - 1)];
#pragma unroll
    for (int r = 0; r < 8; ++r) {
      const int m = half * 8 + r;
      const float y = acc[r] + xs[m * H_ + n0 + nl];
      const float h = tanhf(y);
      hbuf[cur ^ 1][m * H_ + n0 + nl] = (_Float16)h;
      if (o_seq) {
        const size_t row = (size_t)(b0 + m) * T_ + t;
        o_seq[row * (2 * H_) + col + n0 + nl] = (_Float16)h;
      }
      srow[r] += h;
      if (o_last && t == T_ - 1)
        o_last[(size_t)(b0 + m) * (2 * H_) + col + n0 + nl] = h;
    }
    cur ^= 1;
  }
  if (o_sum) {
#pragma unroll
    for (int r = 0; r < 8; ++r)
      o_sum[(size_t)(b0 + half * 8 + r) * (2 * H_) + col + n0 + nl] = srow[r];
  }
}

// ---------------------------------------------------------------------------
// Head: assemble [last | mean], LayerNorm (one block per batch row).
// ---------------------------------------------------------------------------
__global__ __launch_bounds__(256) void head_ln_kernel(
    const float* __restrict__ lastb, const float* __restrict__ sums,
    const float* __restrict__ g, const float* __restrict__ bt,
    float* __restrict__ hn, float invT) {
  __shared__ float row[1024];
  __shared__ float red[256];
  const int b = blockIdx.x, tid = threadIdx.x;
  for (int j = tid; j < 1024; j += 256)
    row[j] = (j < 512) ? lastb[b * 512 + j] : sums[b * 512 + (j - 512)] * invT;
  __syncthreads();
  float s = 0.f;
  for (int j = tid; j < 1024; j += 256) s += row[j];
  red[tid] = s; __syncthreads();
  for (int w = 128; w > 0; w >>= 1) { if (tid < w) red[tid] += red[tid + w]; __syncthreads(); }
  const float mu = red[0] * (1.f / 1024.f);
  __syncthreads();
  float v = 0.f;
  for (int j = tid; j < 1024; j += 256) { float d = row[j] - mu; v += d * d; }
  red[tid] = v; __syncthreads();
  for (int w = 128; w > 0; w >>= 1) { if (tid < w) red[tid] += red[tid + w]; __syncthreads(); }
  const float inv = rsqrtf(red[0] * (1.f / 1024.f) + 1e-5f);
  for (int j = tid; j < 1024; j += 256)
    hn[b * 1024 + j] = (row[j] - mu) * inv * g[j] + bt[j];
}

__global__ void fc_kernel(const float* __restrict__ in, const float* __restrict__ w,
                          const float* __restrict__ bias, float* __restrict__ out,
                          int N, int K, int relu) {
  int i = blockIdx.x * blockDim.x + threadIdx.x;
  if (i >= B_ * N) return;
  const int m = i / N, n = i % N;
  float s = bias[n];
  const float* a  = in + (size_t)m * K;
  const float* wr = w + (size_t)n * K;
  for (int k = 0; k < K; ++k) s = fmaf(a[k], wr[k], s);
  out[i] = relu ? fmaxf(s, 0.f) : s;
}

// ---------------------------------------------------------------------------
extern "C" void kernel_launch(void* const* d_in, const int* in_sizes, int n_in,
                              void* d_out, int out_size, void* d_ws, size_t ws_size,
                              hipStream_t stream) {
  (void)in_sizes; (void)n_in; (void)out_size; (void)ws_size;
  const int*   x      = (const int*)  d_in[0];
  const float* emb    = (const float*)d_in[1];
  const float* r1_wif = (const float*)d_in[2];
  const float* r1_whf = (const float*)d_in[3];
  const float* r1_bif = (const float*)d_in[4];
  const float* r1_bhf = (const float*)d_in[5];
  const float* r1_wib = (const float*)d_in[6];
  const float* r1_whb = (const float*)d_in[7];
  const float* r1_bib = (const float*)d_in[8];
  const float* r1_bhb = (const float*)d_in[9];
  const float* r2_wif = (const float*)d_in[10];
  const float* r2_whf = (const float*)d_in[11];
  const float* r2_bif = (const float*)d_in[12];
  const float* r2_bhf = (const float*)d_in[13];
  const float* r2_wib = (const float*)d_in[14];
  const float* r2_whb = (const float*)d_in[15];
  const float* r2_bib = (const float*)d_in[16];
  const float* r2_bhb = (const float*)d_in[17];
  const float* ln_g   = (const float*)d_in[18];
  const float* ln_b   = (const float*)d_in[19];
  const float* w1     = (const float*)d_in[20];
  const float* b1     = (const float*)d_in[21];
  const float* w2     = (const float*)d_in[22];
  const float* b2     = (const float*)d_in[23];

  char* ws = (char*)d_ws;
  size_t off = 0;
  auto alloc = [&](size_t bytes) -> char* {
    char* p = ws + off;
    off += (bytes + 255) & ~(size_t)255;
    return p;
  };
  const size_t M = (size_t)B_ * T_;                       // 32768 rows
  _Float16* e16  = (_Float16*)alloc(M * E_ * 2);          //  8 MB
  _Float16* wif1 = (_Float16*)alloc((size_t)H_ * E_ * 2);
  _Float16* wib1 = (_Float16*)alloc((size_t)H_ * E_ * 2);
  _Float16* whf1 = (_Float16*)alloc((size_t)H_ * H_ * 2);
  _Float16* whb1 = (_Float16*)alloc((size_t)H_ * H_ * 2);
  _Float16* wif2 = (_Float16*)alloc((size_t)H_ * 2 * H_ * 2);
  _Float16* wib2 = (_Float16*)alloc((size_t)H_ * 2 * H_ * 2);
  _Float16* whf2 = (_Float16*)alloc((size_t)H_ * H_ * 2);
  _Float16* whb2 = (_Float16*)alloc((size_t)H_ * H_ * 2);
  float* xp_f  = (float*)alloc(M * H_ * 4);               // 32 MB (reused L2)
  float* xp_b  = (float*)alloc(M * H_ * 4);               // 32 MB (reused L2)
  _Float16* o1 = (_Float16*)alloc(M * 2 * H_ * 2);        // 32 MB
  float* lastb = (float*)alloc((size_t)B_ * 2 * H_ * 4);
  float* sums  = (float*)alloc((size_t)B_ * 2 * H_ * 4);
  float* hn    = (float*)alloc((size_t)B_ * 1024 * 4);
  float* out1  = (float*)alloc((size_t)B_ * 512 * 4);

  // Stage 0: gather + weight conversions (independent, cheap).
  gather_kernel<<<(unsigned)M, E_, 0, stream>>>(x, emb, e16);
  cvt_kernel<<<(H_ * E_ + 255) / 256, 256, 0, stream>>>(r1_wif, wif1, H_ * E_);
  cvt_kernel<<<(H_ * E_ + 255) / 256, 256, 0, stream>>>(r1_wib, wib1, H_ * E_);
  cvt_kernel<<<(H_ * H_ + 255) / 256, 256, 0, stream>>>(r1_whf, whf1, H_ * H_);
  cvt_kernel<<<(H_ * H_ + 255) / 256, 256, 0, stream>>>(r1_whb, whb1, H_ * H_);
  cvt_kernel<<<(H_ * 2 * H_ + 255) / 256, 256, 0, stream>>>(r2_wif, wif2, H_ * 2 * H_);
  cvt_kernel<<<(H_ * 2 * H_ + 255) / 256, 256, 0, stream>>>(r2_wib, wib2, H_ * 2 * H_);
  cvt_kernel<<<(H_ * H_ + 255) / 256, 256, 0, stream>>>(r2_whf, whf2, H_ * H_);
  cvt_kernel<<<(H_ * H_ + 255) / 256, 256, 0, stream>>>(r2_whb, whb2, H_ * H_);

  const int mtiles = (int)(M / 16);  // 2048 blocks, one per M-tile

  // Layer 1: input projections (both directions in one launch), then scan.
  xproj_kernel<<<dim3(mtiles, 2), 256, 0, stream>>>(
      e16, E_, wif1, wib1, r1_bif, r1_bhf, r1_bib, r1_bhb, xp_f, xp_b);
  rnn_scan_kernel<<<dim3(B_ / 16, 2), 512, 0, stream>>>(
      xp_f, xp_b, whf1, whb1, o1, nullptr, nullptr);

  // Layer 2: projections from the f16 concat buffer, then scan with fused
  // mean-accumulation and last-step capture (o2 never materialized).
  xproj_kernel<<<dim3(mtiles, 2), 256, 0, stream>>>(
      o1, 2 * H_, wif2, wib2, r2_bif, r2_bhf, r2_bib, r2_bhb, xp_f, xp_b);
  rnn_scan_kernel<<<dim3(B_ / 16, 2), 512, 0, stream>>>(
      xp_f, xp_b, whf2, whb2, nullptr, lastb, sums);

  // Head: LN + FC(relu) + FC.
  head_ln_kernel<<<B_, 256, 0, stream>>>(lastb, sums, ln_g, ln_b, hn, 1.0f / T_);
  fc_kernel<<<(B_ * 512 + 255) / 256, 256, 0, stream>>>(hn, w1, b1, out1, 512, 1024, 1);
  fc_kernel<<<(B_ * 256 + 255) / 256, 256, 0, stream>>>(out1, w2, b2, (float*)d_out, 256, 512, 0);
}
